// DampedIMEX1Layer_22368189677865
// MI455X (gfx1250) — matrix operand order repaired and checked by reading, not compiled, not run on hardware
//
#include <hip/hip_runtime.h>

// DampedIMEX1Layer for MI455X (gfx1250, wave32).
// Split-bf16 WMMA GEMMs (f32 = bf16_hi + bf16_lo; hi*hi + hi*lo + lo*hi with
// f32 accumulate) -> ~f32 accuracy at bf16-WMMA rate (16x16x32, 16384
// FLOP/instr vs 2048 for f32 16x16x4).
// Bu is never stored to HBM: each 64-row chunk's Bu GEMM goes to LDS, is
// scanned there (chunk-parallel scan with M^64 carry fixup), overwritten
// in-place by ys, and fed to the output WMMA GEMM from LDS.
// HBM traffic ~= 2x read x (64MB) + write out (32MB) ~ 100MB.
// Workspace need: ~4.5 MB.

#define BSZv   8
#define LSEQ   8192
#define PDIM   128
#define HDIM   128
#define CHUNK  64
#define NCHUNK (LSEQ / CHUNK)   // 128

typedef float  v4f  __attribute__((ext_vector_type(4)));
typedef float  v8f  __attribute__((ext_vector_type(8)));
typedef __bf16 v16bf __attribute__((ext_vector_type(16)));

static constexpr size_t NST     = (size_t)BSZv * 2 * NCHUNK * PDIM;  // 262144
static constexpr size_t F_CONST = 0;                                  // 1280 used
static constexpr size_t F_LF1   = 2048;
static constexpr size_t F_LF2   = F_LF1 + NST;
static constexpr size_t F_CA1   = F_LF2 + NST;
static constexpr size_t F_CA2   = F_CA1 + NST;
static constexpr size_t F_BF    = F_CA2 + NST;        // bf16 planes after this
static constexpr int    PLANE   = PDIM * HDIM;         // 16384

__device__ __forceinline__ int sw(int row, int k) { return (k + (row << 2)) & 127; }

// ---------------------------------------------------------------- constants
__global__ void k_consts(const float* __restrict__ A_diag,
                         const float* __restrict__ G_diag,
                         const float* __restrict__ dt,
                         float* __restrict__ cons) {
  int p = threadIdx.x;
  float dts = 1.0f / (1.0f + __expf(-dt[p]));
  float A = fmaxf(A_diag[p], 0.0f);
  float G = fmaxf(G_diag[p], 0.0f);          // _project_G #1 just relus G
  float dt2 = fmaxf(dts * dts, 1e-6f);
  float sq = sqrtf(1.0f + dts * G);
  float A_low  = (2.0f + dts * G - 2.0f * sq) / dt2;
  float A_high = (2.0f + dts * G + 2.0f * sq) / dt2;
  float A_fin = A_low + fmaxf(A - A_low, 0.0f) - fmaxf(A - A_high, 0.0f);
  float S = 1.0f + dts * G;
  float m11 = 1.0f / S;
  float m21 = dts / S;
  float m12 = -m21 * A_fin;
  float m22 = 1.0f - dts * dts / S * A_fin;
  cons[p]       = m11;
  cons[128 + p] = m12;
  cons[256 + p] = m21;
  cons[384 + p] = m22;
  cons[512 + p] = m21;              // f1c = dts/S
  cons[640 + p] = dts * dts / S;    // f2c
  // Mc = M^CHUNK, CHUNK = 2^6 -> 6 squarings
  float a = m11, b = m12, c = m21, d = m22;
  for (int i = 0; i < 6; ++i) {
    float na = a * a + b * c, nb = a * b + b * d;
    float nc = c * a + d * c, nd = c * b + d * d;
    a = na; b = nb; c = nc; d = nd;
  }
  cons[768 + p]  = a;
  cons[896 + p]  = b;
  cons[1024 + p] = c;
  cons[1152 + p] = d;
}

// ---------------- deinterleave B/C + split each f32 into bf16 hi/lo planes
// bpl: [BHI_RE | BLO_RE | BHI_IM | BLO_IM], each PLANE, layout [p][h]
// cpl: [CHI_RE | CLO_RE | CHI_IM | CLO_IM], each PLANE, layout [h][p]
__global__ void k_prep(const float* __restrict__ B, const float* __restrict__ C,
                       __bf16* __restrict__ bpl, __bf16* __restrict__ cpl) {
  int i = blockIdx.x * 256 + threadIdx.x;      // 16384 threads
  float bre = B[2 * i], bim = B[2 * i + 1];
  __bf16 h;
  h = (__bf16)bre; bpl[i] = h;              bpl[PLANE + i]     = (__bf16)(bre - (float)h);
  h = (__bf16)bim; bpl[2 * PLANE + i] = h;  bpl[3 * PLANE + i] = (__bf16)(bim - (float)h);
  float cre = C[2 * i], cim = C[2 * i + 1];
  h = (__bf16)cre; cpl[i] = h;              cpl[PLANE + i]     = (__bf16)(cre - (float)h);
  h = (__bf16)cim; cpl[2 * PLANE + i] = h;  cpl[3 * PLANE + i] = (__bf16)(cim - (float)h);
}

// ---- shared GEMM phase: Bu[64 l][128 p] (one comp per half-block) -> LDS
// A frag (16-bit A 16x32, contiguous k): lane m holds M=m; elem e -> K =
// e + 16*(lane>=16). B frag: elem e -> K = e + 16*(lane>=16).
__device__ __forceinline__ void bu_gemm_to_lds(const float* __restrict__ x,
                                               const __bf16* __restrict__ bpl,
                                               float (*tile)[CHUNK][128],
                                               int row0blk, int tid) {
  const int lane = tid & 31, wave = tid >> 5;
  const int comp = wave >> 2, lt = wave & 3;
  const int m = lane & 15;
  const int base8 = (lane >> 4) * 8;
  const int khalf = (lane >> 4) * 16;
  const __bf16* bhi = bpl + (comp ? 2 * PLANE : 0);
  const __bf16* blo = bhi + PLANE;
  const int row0 = row0blk + lt * 16;
  const float* xrow = x + (size_t)(row0 + m) * HDIM;

  v8f acc[8] = {};
  for (int kb = 0; kb < HDIM; kb += 32) {
    v16bf ahi, alo;
    const float* pA = xrow + kb + khalf;
    const float* pB = xrow + kb + khalf + 8;
#pragma unroll
    for (int e = 0; e < 16; ++e) {
      float xv = (e < 8) ? pA[e] : pB[e - 8];
      __bf16 hh = (__bf16)xv;
      ahi[e] = hh;
      alo[e] = (__bf16)(xv - (float)hh);
    }
#pragma unroll
    for (int t = 0; t < 8; ++t) {
      const int n = t * 16 + m;                       // p
      const v16bf bh = *(const v16bf*)(bhi + (size_t)n * HDIM + kb + khalf);
      const v16bf bl = *(const v16bf*)(blo + (size_t)n * HDIM + kb + khalf);
      acc[t] = __builtin_amdgcn_wmma_f32_16x16x32_bf16(false, ahi, false, bh, (short)0, acc[t], false, false);
      acc[t] = __builtin_amdgcn_wmma_f32_16x16x32_bf16(false, ahi, false, bl, (short)0, acc[t], false, false);
      acc[t] = __builtin_amdgcn_wmma_f32_16x16x32_bf16(false, alo, false, bh, (short)0, acc[t], false, false);
    }
  }
  const int rowhi = (lane >> 4) * 8;
#pragma unroll
  for (int t = 0; t < 8; ++t) {
    const int col = t * 16 + m;
#pragma unroll
    for (int r = 0; r < 8; ++r) {
      const int lr = lt * 16 + r + rowhi;
      tile[comp][lr][sw(lr, col)] = acc[t][r];
    }
  }
}

// ------------------------- pass 1: Bu GEMM + local chunk scan (zero init)
__global__ void __launch_bounds__(256) k1(const float* __restrict__ x,
                                          const __bf16* __restrict__ bpl,
                                          const float* __restrict__ cons,
                                          float* __restrict__ lf1,
                                          float* __restrict__ lf2) {
  __shared__ float tile[2][CHUNK][128];   // 64 KB
  bu_gemm_to_lds(x, bpl, tile, blockIdx.x * CHUNK, threadIdx.x);
  __syncthreads();

  const int p = threadIdx.x & 127, cm = threadIdx.x >> 7;
  const float m11 = cons[p],       m12 = cons[128 + p];
  const float m21 = cons[256 + p], m22 = cons[384 + p];
  const float f1c = cons[512 + p], f2c = cons[640 + p];
  float s1 = 0.0f, s2 = 0.0f;
#pragma unroll 4
  for (int i = 0; i < CHUNK; ++i) {
    const float u = tile[cm][i][sw(i, p)];
    const float n1 = fmaf(m11, s1, fmaf(m12, s2, f1c * u));
    const float n2 = fmaf(m21, s1, fmaf(m22, s2, f2c * u));
    s1 = n1; s2 = n2;
  }
  const int b = blockIdx.x >> 7, chunk = blockIdx.x & (NCHUNK - 1);
  const size_t o = ((size_t)(b * 2 + cm) * NCHUNK + chunk) * PDIM + p;
  lf1[o] = s1; lf2[o] = s2;
}

// ------------------------------- sequential carry combine via Mc = M^CHUNK
__global__ void __launch_bounds__(256) k_carry(const float* __restrict__ cons,
                                               const float* __restrict__ lf1,
                                               const float* __restrict__ lf2,
                                               float* __restrict__ ca1,
                                               float* __restrict__ ca2) {
  const int idx  = blockIdx.x * 256 + threadIdx.x;   // 2048 chains
  const int p    = idx & 127;
  const int comp = (idx >> 7) & 1;
  const int b    = idx >> 8;
  const float a = cons[768 + p], bb = cons[896 + p];
  const float c = cons[1024 + p], d = cons[1152 + p];
  const size_t base = ((size_t)(b * 2 + comp) * NCHUNK) * PDIM + p;
  float s1 = 0.0f, s2 = 0.0f;
  for (int cc = 0; cc < NCHUNK; ++cc) {
    const size_t o = base + (size_t)cc * PDIM;
    ca1[o] = s1; ca2[o] = s2;            // carry-in (state before this chunk)
    const float n1 = fmaf(a, s1, fmaf(bb, s2, lf1[o]));
    const float n2 = fmaf(c, s1, fmaf(d,  s2, lf2[o]));
    s1 = n1; s2 = n2;
  }
}

// -- pass 2: Bu GEMM -> LDS, rescan from carry (ys in-place, -im), out GEMM
__global__ void __launch_bounds__(256) k2(const float* __restrict__ x,
                                          const __bf16* __restrict__ bpl,
                                          const __bf16* __restrict__ cpl,
                                          const float* __restrict__ D,
                                          const float* __restrict__ cons,
                                          const float* __restrict__ ca1,
                                          const float* __restrict__ ca2,
                                          float* __restrict__ out) {
  __shared__ float tile[2][CHUNK][128];   // 64 KB: Bu, then ys (re / -im)
  bu_gemm_to_lds(x, bpl, tile, blockIdx.x * CHUNK, threadIdx.x);
  __syncthreads();

  // rescan chunk from carry, overwrite Bu with ys in place
  {
    const int p = threadIdx.x & 127, cm = threadIdx.x >> 7;
    const float m11 = cons[p],       m12 = cons[128 + p];
    const float m21 = cons[256 + p], m22 = cons[384 + p];
    const float f1c = cons[512 + p], f2c = cons[640 + p];
    const int b = blockIdx.x >> 7, chunk = blockIdx.x & (NCHUNK - 1);
    const size_t co = ((size_t)(b * 2 + cm) * NCHUNK + chunk) * PDIM + p;
    float s1 = ca1[co], s2 = ca2[co];
#pragma unroll 4
    for (int i = 0; i < CHUNK; ++i) {
      const float u = tile[cm][i][sw(i, p)];
      const float n1 = fmaf(m11, s1, fmaf(m12, s2, f1c * u));
      const float n2 = fmaf(m21, s1, fmaf(m22, s2, f2c * u));
      s1 = n1; s2 = n2;
      tile[cm][i][sw(i, p)] = cm ? -s2 : s2;   // ys, imag pre-negated
    }
  }
  __syncthreads();

  // out[64 x 128] = ys_re*C_re^T - ys_im*C_im^T   (+ D*x at store)
  const int lane = threadIdx.x & 31, wave = threadIdx.x >> 5;
  const int m = lane & 15;
  const int base8 = (lane >> 4) * 8;
  const int khalf = (lane >> 4) * 16;
  const int lt  = wave & 3;                 // l-tile
  const int htg = wave >> 2;                // h-tiles htg + 2*j
  const int row = lt * 16 + m;

  v8f acc[4] = {};
  for (int c2 = 0; c2 < 2; ++c2) {
    const __bf16* chi = cpl + (c2 ? 2 * PLANE : 0);
    const __bf16* clo = chi + PLANE;
    for (int kb = 0; kb < PDIM; kb += 32) {
      v16bf ahi, alo;
#pragma unroll
      for (int j = 0; j < 4; ++j) {       // 4-float chunks, never wrap mod 128
        const int ks = kb + khalf + j * 4;
        const float* pt = &tile[c2][row][sw(row, ks)];
#pragma unroll
        for (int q = 0; q < 4; ++q) {
          const float v = pt[q];
          const __bf16 hh = (__bf16)v;
          ahi[j * 4 + q] = hh;
          alo[j * 4 + q] = (__bf16)(v - (float)hh);
        }
      }
#pragma unroll
      for (int j = 0; j < 4; ++j) {
        const int n = (htg + 2 * j) * 16 + m;     // h
        const v16bf ch = *(const v16bf*)(chi + (size_t)n * PDIM + kb + khalf);
        const v16bf cl = *(const v16bf*)(clo + (size_t)n * PDIM + kb + khalf);
        acc[j] = __builtin_amdgcn_wmma_f32_16x16x32_bf16(false, ahi, false, ch, (short)0, acc[j], false, false);
        acc[j] = __builtin_amdgcn_wmma_f32_16x16x32_bf16(false, ahi, false, cl, (short)0, acc[j], false, false);
        acc[j] = __builtin_amdgcn_wmma_f32_16x16x32_bf16(false, alo, false, ch, (short)0, acc[j], false, false);
      }
    }
  }
  const int rowhi = (lane >> 4) * 8;
#pragma unroll
  for (int j = 0; j < 4; ++j) {
    const int h = (htg + 2 * j) * 16 + m;
    const float dh = D[h];
#pragma unroll
    for (int r = 0; r < 8; ++r) {
      const int rowg = blockIdx.x * CHUNK + lt * 16 + r + rowhi;
      const size_t o = (size_t)rowg * HDIM + h;
      out[o] = acc[j][r] + dh * x[o];
    }
  }
}

// ---------------------------------------------------------------- launcher
extern "C" void kernel_launch(void* const* d_in, const int* in_sizes, int n_in,
                              void* d_out, int out_size, void* d_ws, size_t ws_size,
                              hipStream_t stream) {
  (void)in_sizes; (void)n_in; (void)out_size; (void)ws_size;
  const float* x      = (const float*)d_in[0];
  const float* A_diag = (const float*)d_in[1];
  const float* G_diag = (const float*)d_in[2];
  const float* dt     = (const float*)d_in[3];
  const float* B      = (const float*)d_in[4];
  const float* C      = (const float*)d_in[5];
  const float* D      = (const float*)d_in[6];
  float* out = (float*)d_out;
  float* ws  = (float*)d_ws;

  float* cons = ws + F_CONST;
  float* lf1  = ws + F_LF1;
  float* lf2  = ws + F_LF2;
  float* ca1  = ws + F_CA1;
  float* ca2  = ws + F_CA2;
  __bf16* bpl = (__bf16*)(ws + F_BF);
  __bf16* cpl = bpl + 4 * PLANE;

  k_consts<<<1, 128, 0, stream>>>(A_diag, G_diag, dt, cons);
  k_prep<<<PLANE / 256, 256, 0, stream>>>(B, C, bpl, cpl);
  k1<<<BSZv * NCHUNK, 256, 0, stream>>>(x, bpl, cons, lf1, lf2);
  k_carry<<<(BSZv * 2 * PDIM) / 256, 256, 0, stream>>>(cons, lf1, lf2, ca1, ca2);
  k2<<<BSZv * NCHUNK, 256, 0, stream>>>(x, bpl, cpl, D, cons, ca1, ca2, out);
}
